// _RNN_76321568850654
// MI455X (gfx1250) — compile-verified, matrix operand order checked by reading, not run on hardware
//
#include <hip/hip_runtime.h>

#define HH 256
#define TT 256
#define BBATCH 256
#define NROWS (TT * BBATCH)   // 65536
#define GATES (4 * HH)        // 1024
#define OUTD 48
#define HPAD 264              // padded halves per LDS row (bank-spread)

typedef __attribute__((ext_vector_type(16))) _Float16 v16h;
typedef __attribute__((ext_vector_type(8)))  float    v8f;
typedef __attribute__((ext_vector_type(4)))  int      v4i;

// --- CDNA5 WMMA 16-bit A/B fragment k-mapping (ISA 7.12.2, 16-bit A 16x32) ---
// lane<16: VGPR0..3 -> k=0..7 (2/VGPR), VGPR4..7 -> k=16..23 ; lane>=16: +8
__device__ __forceinline__ int kmap_off(int lane, int e) {
  int v = e >> 1;
  int koff = (v < 4) ? (2 * v) : (16 + 2 * (v - 4));
  return ((lane >> 4) << 3) + koff + (e & 1);
}

// A-fragment: halves 0..7 contiguous at p0, halves 8..15 contiguous at p0+16
__device__ __forceinline__ v16h load_frag32(const _Float16* p0) {
  union { v16h v; v4i q[2]; } u;
  u.q[0] = *(const v4i*)(p0);
  u.q[1] = *(const v4i*)(p0 + 16);
  return u.v;
}

// Pre-swizzled B fragment: 32 contiguous bytes per lane
__device__ __forceinline__ v16h load_fragB(const _Float16* base, int tile, int lane) {
  return *(const v16h*)(base + (((size_t)tile << 5) + (size_t)lane) * 16);
}

__device__ __forceinline__ float sigf(float x)   { return 1.0f / (1.0f + __expf(-x)); }
__device__ __forceinline__ float tanh_f(float x) { return 2.0f * sigf(2.0f * x) - 1.0f; }

__device__ __forceinline__ v8f wmma_f16(v16h a, v16h b, v8f c) {
  return __builtin_amdgcn_wmma_f32_16x16x32_f16(false, a, false, b, (short)0, c,
                                                false, false);
}

// ---------------- preprocessing kernels ----------------

// W is (N x Ksrc) row-major f32; emit B-matrix tiles [kt][nt][lane][16] f16,
// zero-padded for k >= Ksrc.  tile index = kt*(N/16)+nt, B[k][n] = W[n][k].
__global__ void swizzleB_k(const float* __restrict__ W, _Float16* __restrict__ out,
                           int N, int Ksrc, int Ktiles) {
  int idx = blockIdx.x * blockDim.x + threadIdx.x;
  int ntiles = N >> 4;
  int total = Ktiles * ntiles * 32;
  if (idx >= total) return;
  int lane = idx & 31;
  int tile = idx >> 5;
  int nt = tile % ntiles;
  int kt = tile / ntiles;
  int n = (nt << 4) + (lane & 15);
  _Float16* dst = out + ((size_t)tile * 32 + lane) * 16;
  for (int e = 0; e < 16; ++e) {
    int k = (kt << 5) + kmap_off(lane, e);
    float f = (k < Ksrc) ? W[(size_t)n * Ksrc + k] : 0.0f;
    dst[e] = (_Float16)f;
  }
}

__global__ void bias_comb_k(const float* __restrict__ a, const float* __restrict__ b,
                            float* __restrict__ o, int n) {
  int i = blockIdx.x * blockDim.x + threadIdx.x;
  if (i < n) o[i] = a[i] + b[i];
}

// (65536 x 20) f32 -> (65536 x 32) f16, zero padded
__global__ void pad_input_k(const float* __restrict__ x, _Float16* __restrict__ y) {
  int idx = blockIdx.x * blockDim.x + threadIdx.x;
  if (idx >= NROWS * 32) return;
  int r = idx >> 5, d = idx & 31;
  y[idx] = (d < 20) ? (_Float16)x[r * 20 + d] : (_Float16)0.0f;
}

// ---------------- LSTM scan kernel ----------------
// One workgroup (8 wave32) owns a 16-row batch tile for the whole scan.
// Wave w owns hidden j-tiles {2w,2w+1}; it computes gate n-tiles g*16+jt for
// all 4 gates, so the c/h update is wave-local.  h ping-pongs through LDS
// (one barrier per step); gate accumulators start from WMMA inline-0 C and
// the combined bias is added in the elementwise epilogue.
template <int XKT>
__global__ __launch_bounds__(256) void lstm_layer(
    const _Float16* __restrict__ x,    // (S*Bp, XKT*32) f16 row-major
    _Float16* __restrict__ y,          // (S*Bp, 256) f16
    const _Float16* __restrict__ Wih,  // tiles [XKT][64][32][16]
    const _Float16* __restrict__ Whh,  // tiles [8][64][32][16]
    const float* __restrict__ bias,    // 1024 = b_ih + b_hh
    int S, int Bp) {
  __shared__ _Float16 hlds[2][16 * HPAD];
  const int tid = threadIdx.x;
  const int lane = tid & 31;
  const int wave = tid >> 5;
  const int row0 = blockIdx.x << 4;
  const int m = lane & 15;
  const int kb = (lane >> 4) << 3;   // A-fragment k base
  const int rr = (lane >> 4) << 3;   // C-fragment row base

  for (int i = tid; i < 16 * HPAD; i += 256) hlds[0][i] = (_Float16)0.0f;

  int ncol[2];
#pragma unroll
  for (int j = 0; j < 2; ++j) ncol[j] = ((wave << 1) + j) * 16 + m;
  float bsp[4][2];
#pragma unroll
  for (int g = 0; g < 4; ++g)
#pragma unroll
    for (int j = 0; j < 2; ++j) bsp[g][j] = bias[g * 256 + ncol[j]];

  v8f creg[2];
#pragma unroll
  for (int j = 0; j < 2; ++j) creg[j] = (v8f){0, 0, 0, 0, 0, 0, 0, 0};

  __syncthreads();

  const int xK = XKT * 32;
  for (int s = 0; s < S; ++s) {
    const _Float16* xrow = x + ((size_t)s * (size_t)Bp + (size_t)(row0 + m)) * xK;
    v16h xf[XKT];
#pragma unroll
    for (int kt = 0; kt < XKT; ++kt) xf[kt] = load_frag32(xrow + kt * 32 + kb);

    const _Float16* hrow = hlds[s & 1] + m * HPAD;
    v16h hf[8];
#pragma unroll
    for (int kt = 0; kt < 8; ++kt) hf[kt] = load_frag32(hrow + kt * 32 + kb);

    // prefetch next step's x fragments into cache (no LOADcnt cost)
    if (s + 1 < S) {
      const _Float16* xn =
          x + ((size_t)(s + 1) * (size_t)Bp + (size_t)(row0 + m)) * xK;
#pragma unroll
      for (int kt = 0; kt < XKT; ++kt) __builtin_prefetch(xn + kt * 32 + kb, 0, 3);
    }

    float hv[2][8];
#pragma unroll
    for (int j = 0; j < 2; ++j) {
      const int ntbase = (wave << 1) + j;
      v8f acc[4];
#pragma unroll
      for (int g = 0; g < 4; ++g) {
        const int ntile = (g << 4) + ntbase;
        // start from WMMA inline-0 accumulator (bias added in epilogue)
        v8f a = wmma_f16(xf[0], load_fragB(Wih, ntile, lane),
                         (v8f){0, 0, 0, 0, 0, 0, 0, 0});
#pragma unroll
        for (int kt = 1; kt < XKT; ++kt)
          a = wmma_f16(xf[kt], load_fragB(Wih, kt * 64 + ntile, lane), a);
#pragma unroll
        for (int kt = 0; kt < 8; ++kt)
          a = wmma_f16(hf[kt], load_fragB(Whh, kt * 64 + ntile, lane), a);
        acc[g] = a;
      }
#pragma unroll
      for (int r = 0; r < 8; ++r) {
        float iv = acc[0][r] + bsp[0][j];
        float fv = acc[1][r] + bsp[1][j];
        float gv = acc[2][r] + bsp[2][j];
        float ov = acc[3][r] + bsp[3][j];
        float cv = sigf(fv) * creg[j][r] + sigf(iv) * tanh_f(gv);
        creg[j][r] = cv;
        hv[j][r] = sigf(ov) * tanh_f(cv);
      }
    }

    // write h into the other LDS buffer (no reader of it until after barrier)
    _Float16* hnext = hlds[(s + 1) & 1];
#pragma unroll
    for (int j = 0; j < 2; ++j) {
      const int col = ncol[j];
#pragma unroll
      for (int r = 0; r < 8; ++r) {
        _Float16 hh = (_Float16)hv[j][r];
        const int rowi = rr + r;
        hnext[rowi * HPAD + col] = hh;
        y[((size_t)s * (size_t)Bp + (size_t)(row0 + rowi)) * HH + col] = hh;
      }
    }
    __syncthreads();  // h(s+1) visible to all waves
  }
}

// ---------------- head: residual + tanh-MLP + adapter ----------------
__global__ __launch_bounds__(256) void head_kernel(
    const _Float16* __restrict__ xa,    // L11 output
    const _Float16* __restrict__ xb,    // residual (L01 output)
    const _Float16* __restrict__ Wmlp,  // tiles [8][16][32][16]
    const float* __restrict__ bmlp,     // 256
    const _Float16* __restrict__ Wad,   // tiles [8][3][32][16]
    const float* __restrict__ bad,      // 48
    float* __restrict__ out) {          // (65536, 48) f32
  __shared__ _Float16 l0[16 * HPAD];
  __shared__ _Float16 l1[16 * HPAD];
  const int tid = threadIdx.x;
  const int lane = tid & 31;
  const int wave = tid >> 5;
  const size_t row0 = (size_t)blockIdx.x << 4;
  const int m = lane & 15;
  const int kb = (lane >> 4) << 3;
  const int rr = (lane >> 4) << 3;

  {  // residual add into LDS
    const int row = tid >> 4;
    const int col0 = (tid & 15) << 4;
    const size_t g = (row0 + (size_t)row) * HH + col0;
#pragma unroll
    for (int e = 0; e < 16; ++e)
      l0[row * HPAD + col0 + e] = xa[g + e] + xb[g + e];
  }
  __syncthreads();

  {  // z = tanh(xsum @ mlp_w^T + mlp_b)
    const _Float16* arow = l0 + m * HPAD;
    v16h af[8];
#pragma unroll
    for (int kt = 0; kt < 8; ++kt) af[kt] = load_frag32(arow + kt * 32 + kb);
#pragma unroll
    for (int j = 0; j < 2; ++j) {
      const int nt = (wave << 1) + j;
      const int col = nt * 16 + m;
      v8f a = wmma_f16(af[0], load_fragB(Wmlp, nt, lane),
                       (v8f){0, 0, 0, 0, 0, 0, 0, 0});
#pragma unroll
      for (int kt = 1; kt < 8; ++kt)
        a = wmma_f16(af[kt], load_fragB(Wmlp, kt * 16 + nt, lane), a);
      float bv = bmlp[col];
#pragma unroll
      for (int r = 0; r < 8; ++r)
        l1[(rr + r) * HPAD + col] = (_Float16)tanh_f(a[r] + bv);
    }
  }
  __syncthreads();

  if (wave < 3) {  // out = z @ adapter_w^T + adapter_b  (48 cols = 3 n-tiles)
    const _Float16* arow = l1 + m * HPAD;
    v16h af[8];
#pragma unroll
    for (int kt = 0; kt < 8; ++kt) af[kt] = load_frag32(arow + kt * 32 + kb);
    const int nt = wave;
    const int col = nt * 16 + m;
    v8f a = wmma_f16(af[0], load_fragB(Wad, nt, lane),
                     (v8f){0, 0, 0, 0, 0, 0, 0, 0});
#pragma unroll
    for (int kt = 1; kt < 8; ++kt)
      a = wmma_f16(af[kt], load_fragB(Wad, kt * 3 + nt, lane), a);
    float bv = bad[col];
#pragma unroll
    for (int r = 0; r < 8; ++r)
      out[(row0 + (size_t)(rr + r)) * OUTD + col] = a[r] + bv;
  }
}

// ---------------- launcher ----------------
extern "C" void kernel_launch(void* const* d_in, const int* in_sizes, int n_in,
                              void* d_out, int out_size, void* d_ws, size_t ws_size,
                              hipStream_t stream) {
  (void)in_sizes; (void)n_in; (void)out_size; (void)ws_size;
  const float* input = (const float*)d_in[0];
  const float* Wih[4] = {(const float*)d_in[1], (const float*)d_in[5],
                         (const float*)d_in[9], (const float*)d_in[13]};
  const float* Whh[4] = {(const float*)d_in[2], (const float*)d_in[6],
                         (const float*)d_in[10], (const float*)d_in[14]};
  const float* bih[4] = {(const float*)d_in[3], (const float*)d_in[7],
                         (const float*)d_in[11], (const float*)d_in[15]};
  const float* bhh[4] = {(const float*)d_in[4], (const float*)d_in[8],
                         (const float*)d_in[12], (const float*)d_in[16]};
  const float* mlp_w = (const float*)d_in[17];
  const float* mlp_b = (const float*)d_in[18];
  const float* ad_w  = (const float*)d_in[19];
  const float* ad_b  = (const float*)d_in[20];

  char* ws = (char*)d_ws;
  size_t off = 0;
  auto take = [&](size_t bytes) -> char* {
    char* p = ws + off;
    off += (bytes + 255) & ~(size_t)255;
    return p;
  };

  _Float16* x0   = (_Float16*)take((size_t)NROWS * 32 * 2);
  _Float16* bufA = (_Float16*)take((size_t)NROWS * HH * 2);
  _Float16* bufB = (_Float16*)take((size_t)NROWS * HH * 2);
  _Float16* bufC = (_Float16*)take((size_t)NROWS * HH * 2);
  _Float16* WihS[4]; _Float16* WhhS[4]; float* biasC[4];
  for (int l = 0; l < 4; ++l) {
    int kt = (l == 0) ? 1 : 8;
    WihS[l]  = (_Float16*)take((size_t)kt * 64 * 32 * 16 * 2);
    WhhS[l]  = (_Float16*)take((size_t)8 * 64 * 32 * 16 * 2);
    biasC[l] = (float*)take((size_t)GATES * 4);
  }
  _Float16* WmlpS = (_Float16*)take((size_t)8 * 16 * 32 * 16 * 2);
  _Float16* WadS  = (_Float16*)take((size_t)8 * 3 * 32 * 16 * 2);

  auto swz = [&](const float* W, _Float16* o, int N, int Ksrc, int Ktiles) {
    int total = Ktiles * (N >> 4) * 32;
    swizzleB_k<<<(total + 255) / 256, 256, 0, stream>>>(W, o, N, Ksrc, Ktiles);
  };
  swz(Wih[0], WihS[0], GATES, 20, 1);
  for (int l = 1; l < 4; ++l) swz(Wih[l], WihS[l], GATES, HH, 8);
  for (int l = 0; l < 4; ++l) swz(Whh[l], WhhS[l], GATES, HH, 8);
  swz(mlp_w, WmlpS, HH, HH, 8);
  swz(ad_w, WadS, OUTD, HH, 8);
  for (int l = 0; l < 4; ++l)
    bias_comb_k<<<(GATES + 255) / 256, 256, 0, stream>>>(bih[l], bhh[l], biasC[l], GATES);
  pad_input_k<<<(NROWS * 32 + 255) / 256, 256, 0, stream>>>(input, x0);

  // dilated scans: (S, Bp) views are memory no-ops on the flat buffers
  lstm_layer<1><<<BBATCH / 16, 256, 0, stream>>>(x0, bufA, WihS[0], WhhS[0], biasC[0], TT, BBATCH);
  lstm_layer<8><<<(2 * BBATCH) / 16, 256, 0, stream>>>(bufA, bufB, WihS[1], WhhS[1], biasC[1], TT / 2, 2 * BBATCH);
  lstm_layer<8><<<(4 * BBATCH) / 16, 256, 0, stream>>>(bufB, bufC, WihS[2], WhhS[2], biasC[2], TT / 4, 4 * BBATCH);
  lstm_layer<8><<<(8 * BBATCH) / 16, 256, 0, stream>>>(bufC, bufA, WihS[3], WhhS[3], biasC[3], TT / 8, 8 * BBATCH);

  head_kernel<<<NROWS / 16, 256, 0, stream>>>(bufA, bufB, WmlpS, mlp_b, WadS, ad_b, (float*)d_out);
}
